// BayesianKalmanNet_69621419868441
// MI455X (gfx1250) — matrix operand order repaired and checked by reading, not compiled, not run on hardware
//
#include <hip/hip_runtime.h>
#include <hip/hip_bf16.h>

// ---------------- problem constants (from reference) ----------------
#define B_TOT   256
#define T_STEPS 64
#define D_STATE 4
#define O_OBS   2
#define HID     200     // hidden units
#define S_SAMP  10      // ensemble samples
#define BB      4       // batch elements per workgroup -> 64 WGs (latency-bound: maximize WGP parallelism)
#define VROWS   (S_SAMP*BB)   // 40 valid GEMM rows (s,b)
#define ROWS    48            // padded to 3 M-tiles of 16
#define MT      (ROWS/16)     // 3 M-tiles
#define KPAD    224           // K padded to 7*32
#define KT      7             // k-steps of 32
#define NHT     13            // 16-wide tiles covering HID (208)
#define FRAG    512           // bf16 elements per B fragment (32 lanes x 16)

typedef __attribute__((ext_vector_type(16))) __bf16 v16bf;
typedef __attribute__((ext_vector_type(8)))  __bf16 v8bf;
typedef __attribute__((ext_vector_type(8)))  float  v8f;

union AFragU { v16bf v; v8bf h[2]; };

__device__ __forceinline__ v8f zero8() {
  v8f z = {0.f,0.f,0.f,0.f,0.f,0.f,0.f,0.f};
  return z;
}
__device__ __forceinline__ float sigmoidf(float x) { return 1.f/(1.f+__expf(-x)); }
__device__ __forceinline__ float tanh_fast(float x){ return 2.f/(1.f+__expf(-2.f*x)) - 1.f; }

// A-matrix 16x32 bf16 fragment from an LDS buffer with row stride KPAD.
// Contiguous k order used here (not the ISA table order): lane l holds row M = l&15; elements 0..15 cover K = (l>>4)*16..+15
// -> one contiguous 32B chunk (matches B fragment k order).
__device__ __forceinline__ v16bf load_afrag(const __bf16* base, int row, int kt, int lane) {
  const __bf16* p = base + row*KPAD + kt*32 + (lane>>4)*16;
  AFragU f;
  f.h[0] = *(const v8bf*)p;
  f.h[1] = *(const v8bf*)(p + 8);
  return f.v;
}

// B fragments are pre-swizzled: 512 contiguous bf16, lane*16 per lane.
__device__ __forceinline__ v16bf load_bfrag(const __bf16* frag, int lane) {
  return *(const v16bf*)(frag + lane*16);
}

__device__ __forceinline__ v8f wmma_bf16(v16bf a, v16bf b, v8f c) {
  return __builtin_amdgcn_wmma_f32_16x16x32_bf16(false, a, false, b, (short)0, c, false, false);
}

// ---------------- weight pre-swizzle kernels ----------------
// B fragment element (k,n): lane = (n&15) + 16*(k/16), per-lane element e = k%16.
// Gate weights W [3*HID, HID] row-major -> fragments indexed (c, nh, kt).
__global__ void prep_gate_frags(const float* __restrict__ W, __bf16* __restrict__ dst) {
  int idx = blockIdx.x*blockDim.x + threadIdx.x;
  const int total = 3*NHT*KT*FRAG;
  if (idx >= total) return;
  int fi = idx / FRAG;
  int wi = idx % FRAG;
  int lane = wi >> 4;
  int e    = wi & 15;
  int c  = fi / (NHT*KT);
  int r  = fi % (NHT*KT);
  int nh = r / KT;
  int kt = r % KT;
  int n  = lane & 15;
  int k  = ((lane >> 4) << 4) + e;
  int gl = nh*16 + n;        // row within gate chunk
  int k0 = kt*32 + k;        // hidden index
  float v = 0.f;
  if (gl < HID && k0 < HID) v = W[(c*HID + gl)*HID + k0];
  dst[idx] = (__bf16)v;
}

// W_out [8, HID] -> single N-tile (cols 8..15 zero), fragments indexed by kt.
__global__ void prep_out_frags(const float* __restrict__ W, __bf16* __restrict__ dst) {
  int idx = blockIdx.x*blockDim.x + threadIdx.x;
  const int total = KT*FRAG;
  if (idx >= total) return;
  int kt = idx / FRAG;
  int wi = idx % FRAG;
  int lane = wi >> 4;
  int e    = wi & 15;
  int n  = lane & 15;
  int k  = ((lane >> 4) << 4) + e;
  int k0 = kt*32 + k;
  float v = 0.f;
  if (n < D_STATE*O_OBS && k0 < HID) v = W[n*HID + k0];
  dst[idx] = (__bf16)v;
}

// ---------------- main persistent kernel ----------------
__global__ __launch_bounds__(256)
void bkn_kernel(const float* __restrict__ y_seq,
                const float* __restrict__ F_mat, const float* __restrict__ H_mat,
                const float* __restrict__ W_in,  const float* __restrict__ b_in,
                const float* __restrict__ b_ih,  const float* __restrict__ b_hh,
                const float* __restrict__ b_out,
                const __bf16* __restrict__ WihF, const __bf16* __restrict__ WhhF,
                const __bf16* __restrict__ WoutF,
                float* __restrict__ out_xs, float* __restrict__ out_P)
{
  __shared__ __bf16 h_bf[2][ROWS][KPAD];      // GRU state, bf16, double-buffered (rows >= VROWS stay 0)
  __shared__ __bf16 a_bf[16][KPAD];           // input-net activations (rows BB..15 zero pad)
  __shared__ float  gx_lds[BB][3][NHT*16];    // gx (+biases) per gate chunk
  __shared__ float  k_lds[ROWS][D_STATE*O_OBS];
  __shared__ float  xe_lds[S_SAMP][BB][D_STATE];
  __shared__ float  xprev[BB][D_STATE], dxprev[BB][D_STATE];
  __shared__ float  xpred[BB][D_STATE], xfilt[BB][D_STATE];
  __shared__ float  innov[BB][O_OBS];
  __shared__ float  nn_in[BB][6];

  const int tid  = threadIdx.x;
  const int lane = tid & 31;
  const int wv   = tid >> 5;
  const int b0   = blockIdx.x * BB;

  // zero init of state & pads (pads stay zero forever; updates only touch valid rows/cols)
  for (int i = tid; i < 2*ROWS*KPAD; i += 256) ((__bf16*)h_bf)[i] = (__bf16)0.f;
  for (int i = tid; i < 16*KPAD;     i += 256) ((__bf16*)a_bf)[i] = (__bf16)0.f;
  if (tid < BB*D_STATE) { (&xprev[0][0])[tid] = 0.f; (&dxprev[0][0])[tid] = 0.f; }
  __syncthreads();

  for (int t = 0; t < T_STEPS; ++t) {
    const int rd = t & 1, wr = rd ^ 1;

    // ---- A1: x_pred = x_prev @ F^T ----
    if (tid < BB*D_STATE) {
      int lb = tid >> 2, d = tid & 3;
      float s = 0.f;
      #pragma unroll
      for (int j = 0; j < 4; ++j) s += F_mat[d*4+j] * xprev[lb][j];
      xpred[lb][d] = s;
    }
    if (tid == 0 && t+1 < T_STEPS)
      __builtin_prefetch(&y_seq[((size_t)b0*T_STEPS + (t+1))*O_OBS], 0, 1);
    __syncthreads();

    // ---- A2: innovation = y_t - H x_pred ----
    if (tid < BB*O_OBS) {
      int lb = tid >> 1, o = tid & 1;
      float yp = 0.f;
      #pragma unroll
      for (int j = 0; j < 4; ++j) yp += H_mat[o*4+j] * xpred[lb][j];
      innov[lb][o] = y_seq[((size_t)(b0+lb)*T_STEPS + t)*O_OBS + o] - yp;
    }
    __syncthreads();

    // ---- A3: nn_in = [normalize(dx), normalize(innov)] ----
    if (tid < BB) {
      int lb = tid;
      float nd = 0.f;
      #pragma unroll
      for (int j = 0; j < 4; ++j) nd += dxprev[lb][j]*dxprev[lb][j];
      float sc = 1.f / fmaxf(sqrtf(nd), 1e-12f);
      #pragma unroll
      for (int j = 0; j < 4; ++j) nn_in[lb][j] = dxprev[lb][j]*sc;
      float ni = innov[lb][0]*innov[lb][0] + innov[lb][1]*innov[lb][1];
      float si = 1.f / fmaxf(sqrtf(ni), 1e-12f);
      nn_in[lb][4] = innov[lb][0]*si;
      nn_in[lb][5] = innov[lb][1]*si;
    }
    __syncthreads();

    // ---- B: a = relu(nn_in @ W_in^T + b_in)  (K=6, VALU) ----
    for (int idx = tid; idx < BB*HID; idx += 256) {
      int lb = idx / HID, hh = idx % HID;
      float s = b_in[hh];
      #pragma unroll
      for (int j = 0; j < 6; ++j) s += W_in[hh*6+j] * nn_in[lb][j];
      a_bf[lb][hh] = (__bf16)fmaxf(s, 0.f);
    }
    __syncthreads();

    // ---- C: gx = a @ W_ih^T (+ biases), WMMA, shared across samples ----
    for (int tc = wv; tc < 3*NHT; tc += 8) {       // wave-uniform -> EXEC stays full for WMMA
      int c = tc / NHT, nh = tc % NHT;
      v8f acc = zero8();
      const __bf16* wbase = WihF + (size_t)tc * KT * FRAG;
      #pragma unroll
      for (int kt = 0; kt < KT; ++kt) {
        v16bf af = load_afrag(&a_bf[0][0], lane & 15, kt, lane);
        v16bf bf = load_bfrag(wbase + kt*FRAG, lane);
        acc = wmma_bf16(af, bf, acc);
      }
      if (lane < 16) {                              // lanes<16, VGPR j -> C row M=j
        int gcol = nh*16 + lane;
        if (gcol < HID) {
          int g = c*HID + gcol;
          float bias = b_ih[g] + (c < 2 ? b_hh[g] : 0.f);  // n-gate keeps b_hh separate
          #pragma unroll
          for (int j = 0; j < 8; ++j)
            if (j < BB) gx_lds[j][c][gcol] = acc[j] + bias; // only valid batch rows
        }
      }
    }
    __syncthreads();

    // ---- D: gh = h @ W_hh^T (r/z/n fused) + GRU update, WMMA ----
    for (int p = wv; p < MT*NHT; p += 8) {
      int m = p / NHT, nh = p % NHT;
      v8f ar = zero8(), az = zero8(), an = zero8();
      const __bf16* hbase = &h_bf[rd][0][0];
      int row_a = m*16 + (lane & 15);
      #pragma unroll
      for (int kt = 0; kt < KT; ++kt) {
        v16bf af = load_afrag(hbase, row_a, kt, lane);
        v16bf br = load_bfrag(WhhF + ((size_t)(0*NHT+nh)*KT + kt)*FRAG, lane);
        v16bf bz = load_bfrag(WhhF + ((size_t)(1*NHT+nh)*KT + kt)*FRAG, lane);
        v16bf bn = load_bfrag(WhhF + ((size_t)(2*NHT+nh)*KT + kt)*FRAG, lane);
        ar = wmma_bf16(af, br, ar);
        az = wmma_bf16(af, bz, az);
        an = wmma_bf16(af, bn, an);
      }
      int hcol = nh*16 + (lane & 15);
      if (hcol < HID) {
        float bhn = b_hh[2*HID + hcol];
        #pragma unroll
        for (int j = 0; j < 8; ++j) {
          int row = m*16 + j + ((lane >> 4) << 3);  // C layout: lanes>=16 hold M=j+8
          if (row < VROWS) {                        // keep pad rows zero
            int lb  = row & (BB-1);
            float r  = sigmoidf(ar[j] + gx_lds[lb][0][hcol]);
            float z  = sigmoidf(az[j] + gx_lds[lb][1][hcol]);
            float hn = an[j] + bhn;
            float n  = tanh_fast(gx_lds[lb][2][hcol] + r*hn);
            float ho = (float)h_bf[rd][row][hcol];
            h_bf[wr][row][hcol] = (__bf16)((1.f - z)*n + z*ho);
          }
        }
      }
    }
    __syncthreads();

    // ---- E: K = h_new @ W_out^T + b_out, WMMA ----
    if (wv < MT) {
      int m = wv;
      v8f acc = zero8();
      int row_a = m*16 + (lane & 15);
      #pragma unroll
      for (int kt = 0; kt < KT; ++kt) {
        v16bf af = load_afrag(&h_bf[wr][0][0], row_a, kt, lane);
        v16bf bf = load_bfrag(WoutF + kt*FRAG, lane);
        acc = wmma_bf16(af, bf, acc);
      }
      int col = lane & 15;
      if (col < D_STATE*O_OBS) {
        #pragma unroll
        for (int j = 0; j < 8; ++j) {
          int row = m*16 + j + ((lane >> 4) << 3);
          k_lds[row][col] = acc[j] + b_out[col];
        }
      }
    }
    __syncthreads();

    // ---- F: x_ens = x_pred + K @ innov ----
    for (int idx = tid; idx < VROWS*D_STATE; idx += 256) {
      int row = idx >> 2, d = idx & 3;
      int s = row / BB, lb = row & (BB-1);
      xe_lds[s][lb][d] = xpred[lb][d]
                       + k_lds[row][2*d+0]*innov[lb][0]
                       + k_lds[row][2*d+1]*innov[lb][1];
    }
    __syncthreads();

    // ---- G: ensemble mean, write xs, carry x/dx ----
    if (tid < BB*D_STATE) {
      int lb = tid >> 2, d = tid & 3;
      float s = 0.f;
      #pragma unroll
      for (int k = 0; k < S_SAMP; ++k) s += xe_lds[k][lb][d];
      float xf = s * (1.f / (float)S_SAMP);
      xfilt[lb][d] = xf;
      out_xs[((size_t)(b0+lb)*T_STEPS + t)*D_STATE + d] = xf;
      dxprev[lb][d] = xf - xpred[lb][d];
      xprev[lb][d]  = xf;
    }
    __syncthreads();

    // ---- H: ensemble covariance P ----
    if (tid < BB*D_STATE*D_STATE) {
      int lb = tid >> 4, de = tid & 15, d = de >> 2, e = de & 3;
      float s = 0.f;
      float md = xfilt[lb][d], me = xfilt[lb][e];
      #pragma unroll
      for (int k = 0; k < S_SAMP; ++k)
        s += (xe_lds[k][lb][d]-md) * (xe_lds[k][lb][e]-me);
      out_P[(((size_t)(b0+lb)*T_STEPS + t)*D_STATE + d)*D_STATE + e] = s * (1.f/(float)S_SAMP);
    }
    __syncthreads();
  }
}

// ---------------- launch ----------------
extern "C" void kernel_launch(void* const* d_in, const int* in_sizes, int n_in,
                              void* d_out, int out_size, void* d_ws, size_t ws_size,
                              hipStream_t stream) {
  const float* y_seq = (const float*)d_in[0];
  const float* F_mat = (const float*)d_in[1];
  const float* H_mat = (const float*)d_in[2];
  const float* W_in  = (const float*)d_in[3];
  const float* b_in  = (const float*)d_in[4];
  const float* W_ih  = (const float*)d_in[5];
  const float* W_hh  = (const float*)d_in[6];
  const float* b_ih  = (const float*)d_in[7];
  const float* b_hh  = (const float*)d_in[8];
  const float* W_out = (const float*)d_in[9];
  const float* b_out = (const float*)d_in[10];
  (void)in_sizes; (void)n_in; (void)out_size;

  // workspace: pre-swizzled bf16 WMMA B-fragments (~553 KB)
  __bf16* ws    = (__bf16*)d_ws;
  __bf16* WihF  = ws;
  __bf16* WhhF  = ws + (size_t)3*NHT*KT*FRAG;
  __bf16* WoutF = ws + (size_t)2*3*NHT*KT*FRAG;
  (void)ws_size;

  const int nGate = 3*NHT*KT*FRAG;
  prep_gate_frags<<<(nGate+255)/256, 256, 0, stream>>>(W_ih, WihF);
  prep_gate_frags<<<(nGate+255)/256, 256, 0, stream>>>(W_hh, WhhF);
  const int nOut = KT*FRAG;
  prep_out_frags<<<(nOut+255)/256, 256, 0, stream>>>(W_out, WoutF);

  float* out_xs = (float*)d_out;
  float* out_P  = out_xs + (size_t)B_TOT*T_STEPS*D_STATE;
  bkn_kernel<<<B_TOT/BB, 256, 0, stream>>>(y_seq, F_mat, H_mat, W_in, b_in,
                                           b_ih, b_hh, b_out,
                                           WihF, WhhF, WoutF, out_xs, out_P);
}
